// GraphullereneGNN_81501299409277
// MI455X (gfx1250) — matrix operand order repaired and checked by reading, not compiled, not run on hardware
//
#include <hip/hip_runtime.h>

#define F_IN 16
#define D    128
#define H    8
#define C    16
#define NEG_SLOPE 0.2f
#define BN_EPS 1e-5f
#define TPB 256
#define RT_PER_BLK 4
#define BSTRIDE 136   // padded LDS row stride in halves (272B): conflict-free, 16B aligned

typedef __attribute__((ext_vector_type(16))) _Float16 v16h;
typedef __attribute__((ext_vector_type(8)))  _Float16 v8h;
typedef __attribute__((ext_vector_type(8)))  float    v8f;
typedef __attribute__((ext_vector_type(4)))  unsigned int u32x4;
typedef __attribute__((ext_vector_type(8)))  int      i32x8;
typedef __attribute__((ext_vector_type(4)))  int      i32x4;

__device__ __forceinline__ void atomicMaxF(float* addr, float val) {
    unsigned int* ua = reinterpret_cast<unsigned int*>(addr);
    unsigned int cur = *ua;
    while (__uint_as_float(cur) < val) {
        unsigned int prev = atomicCAS(ua, cur, __float_as_uint(val));
        if (prev == cur) break;
        cur = prev;
    }
}

// ---------------- generic fill ----------------
__global__ void k_fill(float* __restrict__ p, float v, long long n) {
    long long i = (long long)blockIdx.x * blockDim.x + threadIdx.x;
    if (i < n) p[i] = v;
}

// ---------------- input projection: h = relu(x @ Wp + bp) ----------------
__global__ void k_input_proj(const float* __restrict__ x, const float* __restrict__ Wp,
                             const float* __restrict__ bp, float* __restrict__ h, int N) {
    long long idx = (long long)blockIdx.x * blockDim.x + threadIdx.x;
    if (idx >= (long long)N * D) return;
    int n = (int)(idx >> 7);
    int d = (int)(idx & (D - 1));
    float s = bp[d];
#pragma unroll
    for (int k = 0; k < F_IN; ++k) s = fmaf(x[(long long)n * F_IN + k], Wp[k * D + d], s);
    h[idx] = s > 0.f ? s : 0.f;
}

// ---------------- f32 -> f16 convert ----------------
__global__ void k_cvt_f16(const float* __restrict__ src, _Float16* __restrict__ dst, long long n) {
    long long i = (long long)blockIdx.x * blockDim.x + threadIdx.x;
    if (i < n) dst[i] = (_Float16)src[i];
}

// ---------------- f32 -> f16 convert + transpose (weights): dst[n,k] = src[k,n] ----------------
__global__ void k_cvt_wT(const float* __restrict__ src, _Float16* __restrict__ dst) {
    int i = blockIdx.x * blockDim.x + threadIdx.x;
    if (i >= D * D) return;
    int n = i >> 7, k = i & (D - 1);
    dst[i] = (_Float16)src[k * D + n];
}

// ---------------- WMMA GEMM: Cout[N,D] = A[N,D] @ B[D,D], B given transposed (BT[n,k]) ----
// BT (32KB f16) is staged into LDS (row-padded) once per block, via the Tensor Data Mover
// when available. block = 256 = 8 waves; wave w -> 16-col tile w; RT_PER_BLK row tiles/block.
__global__ void k_gemm_wmma(const _Float16* __restrict__ A, const _Float16* __restrict__ BT,
                            float* __restrict__ Cout, int N) {
    __shared__ _Float16 sB[D * BSTRIDE];   // row n: 128 K-halves + 8 pad halves

#if __has_builtin(__builtin_amdgcn_tensor_load_to_lds)
    if (threadIdx.x < 32) {                       // wave 0 issues one TDM op
        // generic shared ptr: low 32 bits are the LDS byte offset (aperture truncation)
        unsigned lds_off = (unsigned)(unsigned long long)(uintptr_t)&sB[0];
        unsigned long long ga = (unsigned long long)(uintptr_t)BT;
        // D# group 0: count=1 | lds_addr | global_addr[56:0] | type=2
        u32x4 g0 = {1u, lds_off, (unsigned)ga,
                    (unsigned)((ga >> 32) & 0x01FFFFFFull) | (2u << 30)};
        // D# group 1: data_size=2B; pad_enable, interval=64 DW (row=256B), amount=4 DW (16B);
        //             tensor_dim0=128; tensor_dim1=128; tile_dim0=128; tile_dim1=128;
        //             tensor_dim0_stride=128 elements; rest 0
        i32x8 g1 = {(int)((1u << 16) | (1u << 20) | (5u << 22) | (3u << 25)),
                    (int)(128u << 16),    // [63:48] tensor_dim0 lo16
                    (int)(128u << 16),    // [95:80] tensor_dim1 lo16 (dim0 hi16 = 0)
                    (int)(128u << 16),    // [127:112] tile_dim0 (dim1 hi16 = 0)
                    (int)128,             // [143:128] tile_dim1; tile_dim2 = 0
                    (int)128,             // [191:160] tensor_dim0_stride lo32
                    0, 0};                // stride hi, tensor_dim1_stride = 0
        i32x4 gz4 = {0, 0, 0, 0};
#if defined(__clang_major__) && __clang_major__ >= 23
        i32x8 gz8 = {0, 0, 0, 0, 0, 0, 0, 0};
        __builtin_amdgcn_tensor_load_to_lds(g0, g1, gz4, gz4, gz8, 0);
#else
        __builtin_amdgcn_tensor_load_to_lds(g0, g1, gz4, gz4, 0);
#endif
#if __has_builtin(__builtin_amdgcn_s_wait_tensorcnt)
        __builtin_amdgcn_s_wait_tensorcnt(0);
#endif
    }
#else
    {
        const v8h* src = (const v8h*)BT;
        for (int i = threadIdx.x; i < D * D / 8; i += blockDim.x) {
            int row = i >> 4;            // 16 v8h chunks per 128-half row
            int chk = i & 15;
            *(v8h*)(sB + row * BSTRIDE + chk * 8) = src[i];
        }
    }
#endif
    __syncthreads();

    const int lane = threadIdx.x & 31;
    const int wave = threadIdx.x >> 5;     // 0..7 -> 16-wide column tile
    const int hi   = lane >> 4;            // lane half 0/1
    const int ml   = lane & 15;
    const int nb   = wave * 16;
    // B fragment source: column nb+ml of B = LDS row nb+ml of BT
    const _Float16* Brow = sB + (nb + ml) * BSTRIDE;

    for (int it = 0; it < RT_PER_BLK; ++it) {
        const int rt = blockIdx.x * RT_PER_BLK + it;
        if (rt * 16 >= N) break;
        int row = rt * 16 + ml;
        if (row >= N) row = N - 1;         // clamp (tail-safe loads)
        const _Float16* Arow = A + (long long)row * D;

        v8f acc = {};
#pragma unroll
        for (int kt = 0; kt < D / 32; ++kt) {
            // A fragment: lane needs K = kt*32+hi*16 .. +15
            // -> two contiguous 16B chunks (global b128 loads)
            const v8h* Ap = (const v8h*)(Arow + kt * 32 + hi * 16);
            v8h a0 = Ap[0];
            v8h a1 = Ap[1];
            v16h a = __builtin_shufflevector(a0, a1, 0, 1, 2, 3, 4, 5, 6, 7,
                                             8, 9, 10, 11, 12, 13, 14, 15);
            // B fragment: K = kt*32 + hi*16 + j, contiguous in BT row
            // -> two aligned ds_load_b128
            const v8h* Bp = (const v8h*)(Brow + kt * 32 + hi * 16);
            v8h b0 = Bp[0];
            v8h b1 = Bp[1];
            v16h b = __builtin_shufflevector(b0, b1, 0, 1, 2, 3, 4, 5, 6, 7,
                                             8, 9, 10, 11, 12, 13, 14, 15);

            acc = __builtin_amdgcn_wmma_f32_16x16x32_f16(
                /*neg_a=*/false, a, /*neg_b=*/false, b,
                /*c_mod=*/(short)0, acc, /*reuse_a=*/false, /*reuse_b=*/false);
        }
        // C/D layout: VGPR v -> M = v + hi*8, N = nb + ml
        if (rt * 16 + 16 <= N) {
            // full tile: unconditional stores (uniform branch, no per-lane exec juggling)
            float* Crow = Cout + (long long)(rt * 16 + hi * 8) * D + nb + ml;
#pragma unroll
            for (int v = 0; v < 8; ++v) Crow[(long long)v * D] = acc[v];
        } else {
#pragma unroll
            for (int v = 0; v < 8; ++v) {
                int r = rt * 16 + hi * 8 + v;
                if (r < N) Cout[(long long)r * D + nb + ml] = acc[v];
            }
        }
    }
}

// ---------------- attention coefficients ----------------
__global__ void k_att_coef(const float* __restrict__ xh, const float* __restrict__ as,
                           const float* __restrict__ ad, float* __restrict__ a_src,
                           float* __restrict__ a_dst, int N) {
    int idx = blockIdx.x * blockDim.x + threadIdx.x;
    if (idx >= N * H) return;
    int n = idx >> 3, hh = idx & 7;
    const float* v = xh + (long long)n * D + hh * C;
    float s = 0.f, t = 0.f;
#pragma unroll
    for (int c = 0; c < C; ++c) {
        s = fmaf(v[c], as[hh * C + c], s);
        t = fmaf(v[c], ad[hh * C + c], t);
    }
    a_src[idx] = s;
    a_dst[idx] = t;
}

__device__ __forceinline__ void edge_st(const int* ei, int e, int E, int& s, int& t) {
    if (e < E) { s = ei[e]; t = ei[E + e]; }
    else       { s = t = e - E; }          // self loops appended
}

// ---------------- pass 1: per-dst max (stable softmax) ----------------
__global__ void k_edge_max(const int* __restrict__ ei, const float* __restrict__ a_src,
                           const float* __restrict__ a_dst, float* __restrict__ m,
                           int E, int ET) {
    int idx = blockIdx.x * blockDim.x + threadIdx.x;
    if (idx >= ET * H) return;
    int e = idx >> 3, hh = idx & 7, s, t;
    edge_st(ei, e, E, s, t);
    float el = a_src[s * H + hh] + a_dst[t * H + hh];
    el = el > 0.f ? el : NEG_SLOPE * el;
    atomicMaxF(&m[t * H + hh], el);
}

// ---------------- pass 2: per-dst exp sum ----------------
__global__ void k_edge_sum(const int* __restrict__ ei, const float* __restrict__ a_src,
                           const float* __restrict__ a_dst, const float* __restrict__ m,
                           float* __restrict__ denom, int E, int ET) {
    int idx = blockIdx.x * blockDim.x + threadIdx.x;
    if (idx >= ET * H) return;
    int e = idx >> 3, hh = idx & 7, s, t;
    edge_st(ei, e, E, s, t);
    float el = a_src[s * H + hh] + a_dst[t * H + hh];
    el = el > 0.f ? el : NEG_SLOPE * el;
    atomicAdd(&denom[t * H + hh], __expf(el - m[t * H + hh]));
}

// ---------------- pass 3: weighted aggregation into acc ----------------
__global__ void k_edge_agg(const int* __restrict__ ei, const float* __restrict__ a_src,
                           const float* __restrict__ a_dst, const float* __restrict__ m,
                           const float* __restrict__ denom, const float* __restrict__ xh,
                           float* __restrict__ acc, int E, int ET) {
    int idx = blockIdx.x * blockDim.x + threadIdx.x;
    if (idx >= ET * H) return;
    int e = idx >> 3, hh = idx & 7, s, t;
    edge_st(ei, e, E, s, t);
    int th = t * H + hh;
    float el = a_src[s * H + hh] + a_dst[th];
    el = el > 0.f ? el : NEG_SLOPE * el;
    float w = __expf(el - m[th]) / denom[th];
    const float* xr = xh + (long long)s * D + hh * C;
    __builtin_prefetch(xr, 0, 1);          // global_prefetch_b8
    float* ar = acc + (long long)t * D + hh * C;
#pragma unroll
    for (int c = 0; c < C; ++c) atomicAdd(&ar[c], w * xr[c]);
}

// ---------------- BatchNorm statistics (one block per feature d, LDS reduce) ----------------
__global__ void k_bn_stats(const float* __restrict__ acc, const float* __restrict__ bg,
                           float* __restrict__ mu, float* __restrict__ rstd, int N) {
    __shared__ float ssum[TPB];
    __shared__ float ssq[TPB];
    int d = blockIdx.x;
    float b = bg[d];
    float s = 0.f, q = 0.f;
    for (int n = threadIdx.x; n < N; n += blockDim.x) {
        float v = acc[(long long)n * D + d] + b;
        s += v;
        q = fmaf(v, v, q);
    }
    ssum[threadIdx.x] = s;
    ssq[threadIdx.x]  = q;
    __syncthreads();
    for (int off = TPB / 2; off > 0; off >>= 1) {
        if (threadIdx.x < off) {
            ssum[threadIdx.x] += ssum[threadIdx.x + off];
            ssq[threadIdx.x]  += ssq[threadIdx.x + off];
        }
        __syncthreads();
    }
    if (threadIdx.x == 0) {
        float mean = ssum[0] / (float)N;
        float var  = ssq[0] / (float)N - mean * mean;
        mu[d]   = mean;
        rstd[d] = rsqrtf(var + BN_EPS);
    }
}

// ---------------- BN apply + ReLU, in place on acc ----------------
__global__ void k_bn_apply(float* __restrict__ hio, const float* __restrict__ bg,
                           const float* __restrict__ mu, const float* __restrict__ rstd,
                           const float* __restrict__ gam, const float* __restrict__ bet, int N) {
    long long idx = (long long)blockIdx.x * blockDim.x + threadIdx.x;
    if (idx >= (long long)N * D) return;
    int d = (int)(idx & (D - 1));
    float v = hio[idx] + bg[d];
    v = (v - mu[d]) * rstd[d] * gam[d] + bet[d];
    hio[idx] = v > 0.f ? v : 0.f;
}

// ---------------- global mean pool (sums + counts) ----------------
__global__ void k_pool(const float* __restrict__ h, const int* __restrict__ batch,
                       float* __restrict__ pooled, float* __restrict__ counts, int N) {
    long long idx = (long long)blockIdx.x * blockDim.x + threadIdx.x;
    if (idx >= (long long)N * D) return;
    int n = (int)(idx >> 7);
    int d = (int)(idx & (D - 1));
    int g = batch[n];
    atomicAdd(&pooled[(long long)g * D + d], h[idx]);
    if (d == 0) atomicAdd(&counts[g], 1.0f);
}

__global__ void k_pool_div(float* __restrict__ pooled, const float* __restrict__ counts, int G) {
    int idx = blockIdx.x * blockDim.x + threadIdx.x;
    if (idx >= G * D) return;
    pooled[idx] /= fmaxf(counts[idx >> 7], 1.0f);
}

// ---------------- 3 output heads: Linear(128->64) -> ReLU -> Linear(64->1) ----------------
__global__ void k_heads(const float* __restrict__ pooled, const float* __restrict__ hW1,
                        const float* __restrict__ hb1, const float* __restrict__ hW2,
                        const float* __restrict__ hb2, float* __restrict__ out, int G) {
    int idx = blockIdx.x * blockDim.x + threadIdx.x;
    if (idx >= 3 * G) return;
    int k = idx / G, g = idx % G;
    const float* p = pooled + (long long)g * D;
    float o = hb2[k];
    for (int mm = 0; mm < D / 2; ++mm) {
        float hs = hb1[k * (D / 2) + mm];
        const float* w1 = hW1 + (long long)k * D * (D / 2) + mm;
#pragma unroll 8
        for (int d = 0; d < D; ++d) hs = fmaf(p[d], w1[d * (D / 2)], hs);
        hs = hs > 0.f ? hs : 0.f;
        o = fmaf(hs, hW2[k * (D / 2) + mm], o);
    }
    out[g * 3 + k] = o;
}

extern "C" void kernel_launch(void* const* d_in, const int* in_sizes, int n_in,
                              void* d_out, int out_size, void* d_ws, size_t ws_size,
                              hipStream_t stream) {
    (void)n_in; (void)ws_size;
    const float* x     = (const float*)d_in[0];
    const int*   ei    = (const int*)  d_in[1];
    const int*   batch = (const int*)  d_in[2];
    const float* Wp    = (const float*)d_in[3];
    const float* bp    = (const float*)d_in[4];
    const float* Wg    = (const float*)d_in[5];
    const float* asrcW = (const float*)d_in[6];
    const float* adstW = (const float*)d_in[7];
    const float* bg    = (const float*)d_in[8];
    const float* gam   = (const float*)d_in[9];
    const float* bet   = (const float*)d_in[10];
    const float* hW1   = (const float*)d_in[11];
    const float* hb1   = (const float*)d_in[12];
    const float* hW2   = (const float*)d_in[13];
    const float* hb2   = (const float*)d_in[14];
    float* out = (float*)d_out;

    const int N  = in_sizes[0] / F_IN;
    const int E  = in_sizes[1] / 2;
    const int L  = in_sizes[5] / (D * D);
    const int G  = out_size / 3;
    const int ET = E + N;

    // ---- workspace carving (bytes; all chunks 16B-aligned multiples) ----
    char* w = (char*)d_ws;
    const long long ND = (long long)N * D;
    float*    h      = (float*)w;       w += ND * sizeof(float);      // layer state / agg acc
    float*    xh     = (float*)w;       w += ND * sizeof(float);      // post-linear features
    _Float16* hhalf  = (_Float16*)w;    w += ND * sizeof(_Float16);
    _Float16* wThalf = (_Float16*)w;    w += (long long)D * D * sizeof(_Float16);
    float*    a_src  = (float*)w;       w += (long long)N * H * sizeof(float);
    float*    a_dst  = (float*)w;       w += (long long)N * H * sizeof(float);
    float*    mbuf   = (float*)w;       w += (long long)N * H * sizeof(float);
    float*    denom  = (float*)w;       w += (long long)N * H * sizeof(float);
    float*    mu     = (float*)w;       w += D * sizeof(float);
    float*    rstd   = (float*)w;       w += D * sizeof(float);
    float*    pooled = (float*)w;       w += (long long)G * D * sizeof(float);
    float*    counts = (float*)w;       w += (long long)G * sizeof(float);

    const int ndBlocks = (int)((ND + TPB - 1) / TPB);
    const int nhBlocks = (N * H + TPB - 1) / TPB;
    const int edBlocks = (ET * H + TPB - 1) / TPB;
    const int gemmBlocks = (N + 16 * RT_PER_BLK - 1) / (16 * RT_PER_BLK);

    // h = relu(x @ Wp + bp)
    k_input_proj<<<ndBlocks, TPB, 0, stream>>>(x, Wp, bp, h, N);

    for (int l = 0; l < L; ++l) {
        k_cvt_f16<<<ndBlocks, TPB, 0, stream>>>(h, hhalf, ND);
        k_cvt_wT<<<(D * D + TPB - 1) / TPB, TPB, 0, stream>>>(Wg + (long long)l * D * D, wThalf);
        k_gemm_wmma<<<gemmBlocks, TPB, 0, stream>>>(hhalf, wThalf, xh, N);
        k_att_coef<<<nhBlocks, TPB, 0, stream>>>(xh, asrcW + l * H * C, adstW + l * H * C,
                                                 a_src, a_dst, N);
        k_fill<<<nhBlocks, TPB, 0, stream>>>(mbuf,  -3.402823466e38f, (long long)N * H);
        k_fill<<<nhBlocks, TPB, 0, stream>>>(denom, 0.f,              (long long)N * H);
        k_fill<<<ndBlocks, TPB, 0, stream>>>(h,     0.f,              ND);   // acc buffer
        k_edge_max<<<edBlocks, TPB, 0, stream>>>(ei, a_src, a_dst, mbuf, E, ET);
        k_edge_sum<<<edBlocks, TPB, 0, stream>>>(ei, a_src, a_dst, mbuf, denom, E, ET);
        k_edge_agg<<<edBlocks, TPB, 0, stream>>>(ei, a_src, a_dst, mbuf, denom, xh, h, E, ET);
        k_bn_stats<<<D, TPB, 0, stream>>>(h, bg + l * D, mu, rstd, N);
        k_bn_apply<<<ndBlocks, TPB, 0, stream>>>(h, bg + l * D, mu, rstd,
                                                 gam + l * D, bet + l * D, N);
    }

    k_fill<<<(int)(((long long)G * D + TPB - 1) / TPB), TPB, 0, stream>>>(pooled, 0.f, (long long)G * D);
    k_fill<<<(G + TPB - 1) / TPB, TPB, 0, stream>>>(counts, 0.f, G);
    k_pool<<<ndBlocks, TPB, 0, stream>>>(h, batch, pooled, counts, N);
    k_pool_div<<<(G * D + TPB - 1) / TPB, TPB, 0, stream>>>(pooled, counts, G);
    k_heads<<<(3 * G + TPB - 1) / TPB, TPB, 0, stream>>>(pooled, hW1, hb1, hW2, hb2, out, G);
}
